// STGI_88553635709431
// MI455X (gfx1250) — matrix operand order repaired and checked by reading, not compiled, not run on hardware
//
#include <hip/hip_runtime.h>
#include <hip/hip_bf16.h>
#include <math.h>

// ---------------------------------------------------------------- types
typedef __bf16 bf16_t;
typedef __attribute__((ext_vector_type(16))) bf16_t v16bf;
typedef __attribute__((ext_vector_type(8)))  float  v8f;

#define SEQ   1024   // N in reference (sequence length for LSTM)
#define BATCH 32     // T in reference (batch for LSTM)
#define MROWS 32768  // T*N graph nodes
#define GFEAT 256    // GCN hidden/out
#define HID   256    // LSTM hidden
#define FIN   32

// ------------------------------------------------- WMMA fragment loaders
// A fragment: 16x32 (MxK) bf16, source is f32 row-major with leading dim ld.
// Layout: lanes 0-15 M=0..15 K{0..15}; lanes 16-31 same M,
// K{16..31} (matches load_b_w contiguous order).
__device__ inline v16bf load_a_f32(const float* base, int ld, int m0, int k0,
                                   int lane) {
  const int half = (lane >> 4) & 1;
  const int m    = m0 + (lane & 15);
  const float* p0 = base + (size_t)m * ld + k0 + half * 16;
  const float* p1 = p0 + 8;
  v16bf a;
#pragma unroll
  for (int e = 0; e < 8; ++e) a[e] = (bf16_t)p0[e];
#pragma unroll
  for (int e = 0; e < 8; ++e) a[8 + e] = (bf16_t)p1[e];
  return a;
}

// B fragment: 32x16 (KxN) bf16 where B[k][n] = W[n][k], W is NxK row-major
// bf16.  Lane layout: n = lane&15, k = (lane<16?0:16)+e  -> contiguous 16
// bf16 = one 32B vector load per lane.
__device__ inline v16bf load_b_w(const bf16_t* W, int ldk, int n0, int k0,
                                 int lane) {
  const int half = (lane >> 4) & 1;
  const int n    = n0 + (lane & 15);
  return *reinterpret_cast<const v16bf*>(W + (size_t)n * ldk + k0 + half * 16);
}

__device__ inline v8f wmma_bf16(v16bf a, v16bf b, v8f c) {
  return __builtin_amdgcn_wmma_f32_16x16x32_bf16(false, a, false, b, (short)0,
                                                 c, false, false);
}

__device__ inline float sigmf(float x) { return 1.0f / (1.0f + __expf(-x)); }

// ---------------------------------------------------------- small kernels
__global__ void cast_bf16_k(const float* __restrict__ s, bf16_t* __restrict__ d,
                            int n) {
  int i = blockIdx.x * blockDim.x + threadIdx.x;
  if (i < n) d[i] = (bf16_t)s[i];
}
__global__ void combine_bias_k(const float* __restrict__ a,
                               const float* __restrict__ b,
                               float* __restrict__ o, int n) {
  int i = blockIdx.x * blockDim.x + threadIdx.x;
  if (i < n) o[i] = a[i] + b[i];
}
__global__ void fill_k(float* __restrict__ p, float v, int n) {
  int i = blockIdx.x * blockDim.x + threadIdx.x;
  if (i < n) p[i] = v;
}
__global__ void deg_k(const int* __restrict__ col, float* __restrict__ deg,
                      int E) {
  int i = blockIdx.x * blockDim.x + threadIdx.x;
  if (i < E) atomicAdd(&deg[col[i]], 1.0f);
}
__global__ void rsqrt_k(const float* __restrict__ deg, float* __restrict__ di,
                        int n) {
  int i = blockIdx.x * blockDim.x + threadIdx.x;
  if (i < n) di[i] = rsqrtf(deg[i]);
}

// ------------------------------------------------------------- WMMA GEMM
// C[M,N] = act(A[M,K](f32) @ W[N,K](bf16)^T + bias).
// Register-blocked: one 32x32 output block per wave (2 M-tiles x 2 N-tiles,
// 4 v8f accumulators).  Each loaded A/B fragment feeds two WMMAs, doubling
// arithmetic intensity vs one-tile-per-wave (GEMMs here are L2-BW bound).
__global__ void gemm_wmma_k(const float* __restrict__ A,
                            const bf16_t* __restrict__ W,
                            const float* __restrict__ bias,
                            float* __restrict__ C, int M, int N, int K,
                            int relu) {
  const int lane = threadIdx.x & 31;
  const int wave = threadIdx.x >> 5;
  const int blk  = blockIdx.x * (blockDim.x >> 5) + wave;
  const int blksN = N >> 5;
  const int blksTotal = (M >> 5) * blksN;
  if (blk >= blksTotal) return;
  const int m0 = (blk / blksN) * 32;
  const int n0 = (blk % blksN) * 32;

  v8f acc[2][2];
#pragma unroll
  for (int mt = 0; mt < 2; ++mt)
#pragma unroll
    for (int nt = 0; nt < 2; ++nt) acc[mt][nt] = (v8f){};

  for (int k0 = 0; k0 < K; k0 += 32) {
    v16bf a0 = load_a_f32(A, K, m0, k0, lane);
    v16bf a1 = load_a_f32(A, K, m0 + 16, k0, lane);
    v16bf b0 = load_b_w(W, K, n0, k0, lane);
    v16bf b1 = load_b_w(W, K, n0 + 16, k0, lane);
    acc[0][0] = wmma_bf16(a0, b0, acc[0][0]);
    acc[0][1] = wmma_bf16(a0, b1, acc[0][1]);
    acc[1][0] = wmma_bf16(a1, b0, acc[1][0]);
    acc[1][1] = wmma_bf16(a1, b1, acc[1][1]);
  }

  const int half = lane >> 4;
  const int lcol = lane & 15;
#pragma unroll
  for (int mt = 0; mt < 2; ++mt) {
#pragma unroll
    for (int nt = 0; nt < 2; ++nt) {
      const int n = n0 + nt * 16 + lcol;
      const float bv = bias ? bias[n] : 0.0f;
#pragma unroll
      for (int r = 0; r < 8; ++r) {
        float v = acc[mt][nt][r] + bv;
        if (relu) v = fmaxf(v, 0.0f);
        C[(size_t)(m0 + mt * 16 + half * 8 + r) * N + n] = v;
      }
    }
  }
}

// --------------------------------------------------------- GCN scatter
// out[col] += dinv[row]*dinv[col] * xw[row]  (F=256, 4 floats per thread).
// Random-row gather: prefetch the source line (global_prefetch_b8) before
// the dependent loads/atomics.
__global__ void gcn_scatter_k(const int* __restrict__ row,
                              const int* __restrict__ col,
                              const float* __restrict__ dinv,
                              const float* __restrict__ xw,
                              float* __restrict__ out, int E) {
  long long idx = (long long)blockIdx.x * blockDim.x + threadIdx.x;
  long long total = (long long)E * 64;
  if (idx >= total) return;
  int e  = (int)(idx >> 6);
  int f4 = (int)(idx & 63) << 2;
  int r = row[e], c = col[e];
  const float* src = xw + (size_t)r * GFEAT + f4;
  float* dst = out + (size_t)c * GFEAT + f4;
  __builtin_prefetch(src, 0, 3);
  float norm = dinv[r] * dinv[c];
  atomicAdd(dst + 0, norm * src[0]);
  atomicAdd(dst + 1, norm * src[1]);
  atomicAdd(dst + 2, norm * src[2]);
  atomicAdd(dst + 3, norm * src[3]);
}

// out = relu(out + xw*dinv^2 + bias)
__global__ void gcn_final_k(float* __restrict__ out,
                            const float* __restrict__ xw,
                            const float* __restrict__ dinv,
                            const float* __restrict__ bias, int n) {
  int i = blockIdx.x * blockDim.x + threadIdx.x;
  if (i >= n) return;
  int m = i >> 8, f = i & 255;
  float di = dinv[m];
  out[i] = fmaxf(out[i] + xw[i] * di * di + bias[f], 0.0f);
}

// ------------------------------------------------ persistent BiLSTM scan
// One block per direction (blockIdx.x: 0=fwd,1=bwd), 512 threads = 16 waves.
// Wave w owns hidden units [w*16, w*16+16): all 4 gate tiles for both batch
// halves live in that wave, so the c-state update is lane-local.  h lives in
// LDS (32x256 f32); c lives in registers.
__global__ void __launch_bounds__(512, 1)
lstm_scan_k(const float* __restrict__ x, int in_d,
            const bf16_t* __restrict__ Wih_f, const bf16_t* __restrict__ Whh_f,
            const float* __restrict__ bias_f,
            const bf16_t* __restrict__ Wih_b, const bf16_t* __restrict__ Whh_b,
            const float* __restrict__ bias_b, float* __restrict__ out) {
  const int dir = blockIdx.x;  // also reverse flag
  const bf16_t* Wih  = dir ? Wih_b  : Wih_f;
  const bf16_t* Whh  = dir ? Whh_b  : Whh_f;
  const float*  bias = dir ? bias_b : bias_f;

  __shared__ __align__(32) float hbuf[BATCH * HID];  // 32 KB

  const int lane = threadIdx.x & 31;
  const int wave = threadIdx.x >> 5;
  const int half = lane >> 4;
  const int lcol = lane & 15;
  const int j0   = wave * 16;       // this wave's hidden-unit base
  const int j    = j0 + lcol;       // this lane's hidden unit

  for (int i = threadIdx.x; i < BATCH * HID; i += blockDim.x) hbuf[i] = 0.0f;
  __syncthreads();

  const float bi = bias[0 * HID + j];
  const float bf = bias[1 * HID + j];
  const float bg = bias[2 * HID + j];
  const float bo = bias[3 * HID + j];

  float cst[2][8];   // c state: m = mt*16 + half*8 + r
#pragma unroll
  for (int mt = 0; mt < 2; ++mt)
#pragma unroll
    for (int r = 0; r < 8; ++r) cst[mt][r] = 0.0f;

  for (int s = 0; s < SEQ; ++s) {
    const int t = dir ? (SEQ - 1 - s) : s;

    v8f acc[2][4];   // [m-tile][gate]
#pragma unroll
    for (int mt = 0; mt < 2; ++mt)
#pragma unroll
      for (int g = 0; g < 4; ++g) acc[mt][g] = (v8f){};

    // ---- input projection: gates += x_t @ Wih^T
    const float* xt = x + (size_t)t * in_d;          // batch stride SEQ*in_d
    for (int k0 = 0; k0 < in_d; k0 += 32) {
      v16bf a0 = load_a_f32(xt, SEQ * in_d, 0,  k0, lane);
      v16bf a1 = load_a_f32(xt, SEQ * in_d, 16, k0, lane);
#pragma unroll
      for (int g = 0; g < 4; ++g) {
        v16bf b = load_b_w(Wih, in_d, g * HID + j0, k0, lane);
        acc[0][g] = wmma_bf16(a0, b, acc[0][g]);
        acc[1][g] = wmma_bf16(a1, b, acc[1][g]);
      }
    }
    // ---- recurrence: gates += h @ Whh^T   (h read from LDS)
    for (int k0 = 0; k0 < HID; k0 += 32) {
      v16bf a0 = load_a_f32(hbuf, HID, 0,  k0, lane);
      v16bf a1 = load_a_f32(hbuf, HID, 16, k0, lane);
#pragma unroll
      for (int g = 0; g < 4; ++g) {
        v16bf b = load_b_w(Whh, HID, g * HID + j0, k0, lane);
        acc[0][g] = wmma_bf16(a0, b, acc[0][g]);
        acc[1][g] = wmma_bf16(a1, b, acc[1][g]);
      }
    }

    // ---- gate nonlinearities + state update (lane-local)
    float hnew[2][8];
#pragma unroll
    for (int mt = 0; mt < 2; ++mt) {
#pragma unroll
      for (int r = 0; r < 8; ++r) {
        float iv = sigmf(acc[mt][0][r] + bi);
        float fv = sigmf(acc[mt][1][r] + bf);
        float gv = tanhf(acc[mt][2][r] + bg);
        float ov = sigmf(acc[mt][3][r] + bo);
        float cc = fv * cst[mt][r] + iv * gv;
        cst[mt][r] = cc;
        float hv = ov * tanhf(cc);
        hnew[mt][r] = hv;
        int m = mt * 16 + half * 8 + r;
        out[((size_t)m * SEQ + t) * (2 * HID) + dir * HID + j] = hv;
      }
    }
    __syncthreads();   // all waves done reading old h
#pragma unroll
    for (int mt = 0; mt < 2; ++mt)
#pragma unroll
      for (int r = 0; r < 8; ++r) {
        int m = mt * 16 + half * 8 + r;
        hbuf[m * HID + j] = hnew[mt][r];
      }
    __syncthreads();   // new h visible
  }
}

// ----------------------------------------------- imputation + masked loss
__global__ void impute_loss_k(const float* __restrict__ imp,
                              const float* __restrict__ xm,
                              const float* __restrict__ mask,
                              float* __restrict__ xfinal,
                              float* __restrict__ accum, int n) {
  __shared__ float sa[256], sb[256];
  int i = blockIdx.x * blockDim.x + threadIdx.x;
  float d = 0.0f, msum = 0.0f;
  if (i < n) {
    float im = imp[i], x0 = xm[i], mk = mask[i];
    float diff = im - x0;
    d = mk * diff * diff;
    msum = mk;
    xfinal[i] = (mk != 0.0f) ? x0 : im;
  }
  int tid = threadIdx.x;
  sa[tid] = d; sb[tid] = msum;
  __syncthreads();
  for (int s2 = 128; s2 > 0; s2 >>= 1) {
    if (tid < s2) { sa[tid] += sa[tid + s2]; sb[tid] += sb[tid + s2]; }
    __syncthreads();
  }
  if (tid == 0) {
    atomicAdd(&accum[0], sa[0]);
    atomicAdd(&accum[1], sb[0]);
  }
}
__global__ void write_loss_k(const float* __restrict__ accum,
                             float* __restrict__ loss) {
  loss[0] = accum[0] / (accum[1] + 1e-8f);
}

// ---------------------------------------------------------------- driver
extern "C" void kernel_launch(void* const* d_in, const int* in_sizes, int n_in,
                              void* d_out, int out_size, void* d_ws,
                              size_t ws_size, hipStream_t stream) {
  (void)n_in; (void)out_size; (void)ws_size;
  const float* x_missing = (const float*)d_in[0];
  const int*   edge      = (const int*)d_in[1];
  const float* mask      = (const float*)d_in[2];
  const float* W1 = (const float*)d_in[3];
  const float* b1 = (const float*)d_in[4];
  const float* W2 = (const float*)d_in[5];
  const float* b2 = (const float*)d_in[6];
  const float* Wih[4] = {(const float*)d_in[7],  (const float*)d_in[11],
                         (const float*)d_in[15], (const float*)d_in[19]};
  const float* Whh[4] = {(const float*)d_in[8],  (const float*)d_in[12],
                         (const float*)d_in[16], (const float*)d_in[20]};
  const float* bih[4] = {(const float*)d_in[9],  (const float*)d_in[13],
                         (const float*)d_in[17], (const float*)d_in[21]};
  const float* bhh[4] = {(const float*)d_in[10], (const float*)d_in[14],
                         (const float*)d_in[18], (const float*)d_in[22]};
  const float* Wd = (const float*)d_in[23];
  const float* bd = (const float*)d_in[24];

  const int E = in_sizes[1] / 2;
  const int ihK[4] = {GFEAT, GFEAT, 2 * HID, 2 * HID};

  // ---- workspace carve-up
  char* ws = (char*)d_ws;
  size_t off = 0;
  auto alloc = [&](size_t bytes) -> void* {
    off = (off + 255) & ~(size_t)255;
    void* p = ws + off;
    off += bytes;
    return p;
  };
  bf16_t* W1b = (bf16_t*)alloc(sizeof(bf16_t) * GFEAT * FIN);
  bf16_t* W2b = (bf16_t*)alloc(sizeof(bf16_t) * GFEAT * GFEAT);
  bf16_t* Wdb = (bf16_t*)alloc(sizeof(bf16_t) * FIN * 2 * HID);
  bf16_t* Wihb[4]; bf16_t* Whhb[4]; float* biasc[4];
  for (int d = 0; d < 4; ++d) {
    Wihb[d]  = (bf16_t*)alloc(sizeof(bf16_t) * 4 * HID * ihK[d]);
    Whhb[d]  = (bf16_t*)alloc(sizeof(bf16_t) * 4 * HID * HID);
    biasc[d] = (float*)alloc(sizeof(float) * 4 * HID);
  }
  float* deg  = (float*)alloc(sizeof(float) * MROWS);
  float* dinv = (float*)alloc(sizeof(float) * MROWS);
  float* xw   = (float*)alloc(sizeof(float) * (size_t)MROWS * GFEAT);
  float* g1   = (float*)alloc(sizeof(float) * (size_t)MROWS * GFEAT);
  float* g2   = (float*)alloc(sizeof(float) * (size_t)MROWS * GFEAT);
  float* l0   = (float*)alloc(sizeof(float) * (size_t)MROWS * 2 * HID);
  float* l1   = (float*)alloc(sizeof(float) * (size_t)MROWS * 2 * HID);
  float* imp  = (float*)alloc(sizeof(float) * (size_t)MROWS * FIN);
  float* acc2 = (float*)alloc(sizeof(float) * 2);

  auto grid1 = [](long long n) { return (unsigned)((n + 255) / 256); };

  // ---- weight casts
  cast_bf16_k<<<grid1(GFEAT * FIN), 256, 0, stream>>>(W1, W1b, GFEAT * FIN);
  cast_bf16_k<<<grid1(GFEAT * GFEAT), 256, 0, stream>>>(W2, W2b, GFEAT * GFEAT);
  cast_bf16_k<<<grid1(FIN * 2 * HID), 256, 0, stream>>>(Wd, Wdb, FIN * 2 * HID);
  for (int d = 0; d < 4; ++d) {
    cast_bf16_k<<<grid1(4 * HID * ihK[d]), 256, 0, stream>>>(Wih[d], Wihb[d],
                                                             4 * HID * ihK[d]);
    cast_bf16_k<<<grid1(4 * HID * HID), 256, 0, stream>>>(Whh[d], Whhb[d],
                                                          4 * HID * HID);
    combine_bias_k<<<grid1(4 * HID), 256, 0, stream>>>(bih[d], bhh[d],
                                                       biasc[d], 4 * HID);
  }

  // ---- graph norm (shared by both GCN layers)
  fill_k<<<grid1(MROWS), 256, 0, stream>>>(deg, 1.0f, MROWS);
  deg_k<<<grid1(E), 256, 0, stream>>>(edge + E, deg, E);
  rsqrt_k<<<grid1(MROWS), 256, 0, stream>>>(deg, dinv, MROWS);

  const int nG = MROWS * GFEAT;
  // 32x32 block per wave, 8 waves per block
  const unsigned gemmGrid256 = (MROWS / 32) * (GFEAT / 32) / 8;

  // ---- GCN layer 1
  gemm_wmma_k<<<gemmGrid256, 256, 0, stream>>>(x_missing, W1b, nullptr, xw,
                                               MROWS, GFEAT, FIN, 0);
  fill_k<<<grid1(nG), 256, 0, stream>>>(g1, 0.0f, nG);
  gcn_scatter_k<<<grid1((long long)E * 64), 256, 0, stream>>>(edge, edge + E,
                                                              dinv, xw, g1, E);
  gcn_final_k<<<grid1(nG), 256, 0, stream>>>(g1, xw, dinv, b1, nG);

  // ---- GCN layer 2
  gemm_wmma_k<<<gemmGrid256, 256, 0, stream>>>(g1, W2b, nullptr, xw, MROWS,
                                               GFEAT, GFEAT, 0);
  fill_k<<<grid1(nG), 256, 0, stream>>>(g2, 0.0f, nG);
  gcn_scatter_k<<<grid1((long long)E * 64), 256, 0, stream>>>(edge, edge + E,
                                                              dinv, xw, g2, E);
  gcn_final_k<<<grid1(nG), 256, 0, stream>>>(g2, xw, dinv, b2, nG);

  // ---- BiLSTM (persistent scan; blockIdx = direction)
  lstm_scan_k<<<2, 512, 0, stream>>>(g2, GFEAT, Wihb[0], Whhb[0], biasc[0],
                                     Wihb[1], Whhb[1], biasc[1], l0);
  lstm_scan_k<<<2, 512, 0, stream>>>(l0, 2 * HID, Wihb[2], Whhb[2], biasc[2],
                                     Wihb[3], Whhb[3], biasc[3], l1);

  // ---- decoder + loss + imputation
  const unsigned decGrid = ((MROWS / 32) * (FIN / 32) + 7) / 8;
  gemm_wmma_k<<<decGrid, 256, 0, stream>>>(l1, Wdb, bd, imp, MROWS, FIN,
                                           2 * HID, 0);
  fill_k<<<1, 256, 0, stream>>>(acc2, 0.0f, 2);
  float* xfinal = (float*)d_out;
  const int nOut = MROWS * FIN;
  impute_loss_k<<<grid1(nOut), 256, 0, stream>>>(imp, x_missing, mask, xfinal,
                                                 acc2, nOut);
  write_loss_k<<<1, 1, 0, stream>>>(acc2, xfinal + nOut);
}
